// MMFP4FusedExpert_30915174596904
// MI455X (gfx1250) — matrix operand order repaired and checked by reading, not compiled, not run on hardware
//
#include <hip/hip_runtime.h>

typedef __attribute__((ext_vector_type(16))) _Float16 v16h;
typedef __attribute__((ext_vector_type(8)))  _Float16 v8h;
typedef __attribute__((ext_vector_type(2)))  _Float16 h2;
typedef __attribute__((ext_vector_type(8)))  float    v8f;

#define KB_ 32
#define KH_ 4096
#define KI_ 2048
#define KN2_ (2 * KI_)
#define NWAVE 4  // K-split factor (waves per workgroup)

// FP4(E2M1) -> f16 bit trick:
//   nibble {s,e1,e0,m} relocated to f16 bits (s<<15)|(e1 e0 m <<9)
//   equals true_value * 2^-14 for both normal (e>0) and subnormal (+-0.5, +-0)
//   cases. One packed multiply by (scale * 2^14) finishes dequant in f16.
// The 8 nibbles of one packed word are K-consecutive, which is exactly the
// per-lane element order of the WMMA B fragment -> dequant lands in place.
__device__ __forceinline__ v8h dequant8(unsigned int w, _Float16 s14) {
  h2 sc;
  sc[0] = s14;
  sc[1] = s14;
  v8h r;
#pragma unroll
  for (int j = 0; j < 4; ++j) {
    // nibbles 2j (low half) and 2j+1 (high half), each in bits [3:0] of a half
    unsigned int d = ((w >> (8 * j)) & 0xFu) | (((w >> (8 * j + 4)) & 0xFu) << 16);
    // per-half f16 bits: magnitude -> [11:9], sign -> [15]
    unsigned int bits = ((d & 0x00070007u) << 9) | ((d & 0x00080008u) << 12);
    h2 hv = __builtin_bit_cast(h2, bits);
    h2 hr = hv * sc;  // v_pk_mul_f16
    r[2 * j]     = hr[0];
    r[2 * j + 1] = hr[1];
  }
  return r;
}

__device__ __forceinline__ v16h cat16(v8h lo, v8h hi) {
  return __builtin_shufflevector(lo, hi, 0, 1, 2, 3, 4, 5, 6, 7,
                                 8, 9, 10, 11, 12, 13, 14, 15);
}

// Kernel 1: h = x @ dequant(W_gate_up); act = silu(h_gate) * h_up -> f16 [32, KI_]
// 4 waves per workgroup, each accumulating a disjoint quarter of K, then an
// LDS elementwise reduction of the f32 C fragments (identical (lane,vgpr)
// mapping across waves). grid = KI_/16 = 128 -> 512 concurrent waves.
__global__ __launch_bounds__(32 * NWAVE)
void fp4_gateup_swiglu_kernel(const _Float16* __restrict__ x,
                              const int* __restrict__ wpk,
                              const _Float16* __restrict__ wsc,
                              _Float16* __restrict__ act) {
  __shared__ float red[NWAVE * 4 * 256];  // [wave][acc][lane*8+r] = 16 KB

  const int tid  = threadIdx.x;
  const int wv   = tid >> 5;
  const int lane = tid & 31;
  const int nt   = blockIdx.x;
  const int ng   = nt * 16;        // gate column base in packed [*, 2I]
  const int nu   = KI_ + nt * 16;  // up column base

  const int ncol = lane & 15;              // B column = N index
  const int m    = lane & 15;              // A row within its 16-row tile
  const int koff = (lane < 16) ? 0 : 8;    // A-fragment K sub-offset (ISA layout)
  const int kwb  = (lane < 16) ? 0 : 1;    // B word-row base: K 0..7,16..23 | 8..15,24..31

  v8f accG0 = {}, accG1 = {}, accU0 = {}, accU1 = {};

  const int tsteps = KH_ / 32 / NWAVE;  // 32 K-steps per wave
  const int t0 = wv * tsteps, t1 = t0 + tsteps;
  for (int t = t0; t < t1; ++t) {
    const int kb   = t * 32;
    const int rowp = t * 4 + kwb;  // packed word rows holding this lane's K span

    // Streaming FP4 weight words (read-once -> non-temporal). Word (r, n)
    // holds K = r*8 .. r*8+7 for column n: exactly B-fragment elements.
    unsigned int g0 = (unsigned int)__builtin_nontemporal_load(
        &wpk[(rowp    ) * KN2_ + ng + ncol]);
    unsigned int g1 = (unsigned int)__builtin_nontemporal_load(
        &wpk[(rowp + 2) * KN2_ + ng + ncol]);
    unsigned int u0 = (unsigned int)__builtin_nontemporal_load(
        &wpk[(rowp    ) * KN2_ + nu + ncol]);
    unsigned int u1 = (unsigned int)__builtin_nontemporal_load(
        &wpk[(rowp + 2) * KN2_ + nu + ncol]);
    // scale * 2^14 (exact power-of-two rescale of an f16 scale)
    const _Float16 sg = (_Float16)((float)wsc[(t >> 2) * KN2_ + ng + ncol] * 16384.0f);
    const _Float16 su = (_Float16)((float)wsc[(t >> 2) * KN2_ + nu + ncol] * 16384.0f);
    if (t + 1 < t1) {  // prefetch next K-step's weight stream
      __builtin_prefetch(&wpk[(rowp + 4) * KN2_ + ng + ncol], 0, 1);
      __builtin_prefetch(&wpk[(rowp + 4) * KN2_ + nu + ncol], 0, 1);
    }

    // A fragments straight from global f16 x (L2-resident, 256 KB total).
    const _Float16* xr0 = x + m * KH_ + kb + koff;
    const _Float16* xr1 = x + (m + 16) * KH_ + kb + koff;
    const v16h A0 = cat16(*(const v8h*)xr0, *(const v8h*)(xr0 + 16));
    const v16h A1 = cat16(*(const v8h*)xr1, *(const v8h*)(xr1 + 16));

    // B fragments dequanted directly in registers (no shuffle needed).
    const v16h Bg = cat16(dequant8(g0, sg), dequant8(g1, sg));
    const v16h Bu = cat16(dequant8(u0, su), dequant8(u1, su));

    accG0 = __builtin_amdgcn_wmma_f32_16x16x32_f16(false, A0, false, Bg,
                                                   (short)0, accG0, false, false);
    accG1 = __builtin_amdgcn_wmma_f32_16x16x32_f16(false, A1, false, Bg,
                                                   (short)0, accG1, false, false);
    accU0 = __builtin_amdgcn_wmma_f32_16x16x32_f16(false, A0, false, Bu,
                                                   (short)0, accU0, false, false);
    accU1 = __builtin_amdgcn_wmma_f32_16x16x32_f16(false, A1, false, Bu,
                                                   (short)0, accU1, false, false);
  }

  // Cross-wave K reduction through LDS.
  *(v8f*)&red[(wv * 4 + 0) * 256 + lane * 8] = accG0;
  *(v8f*)&red[(wv * 4 + 1) * 256 + lane * 8] = accG1;
  *(v8f*)&red[(wv * 4 + 2) * 256 + lane * 8] = accU0;
  *(v8f*)&red[(wv * 4 + 3) * 256 + lane * 8] = accU1;
  __syncthreads();

  if (wv < 2) {
    // wave 0 reduces (G0,U0) -> output rows 0..15; wave 1 (G1,U1) -> 16..31.
    const int ag = wv, au = wv + 2;
    v8f G = {}, U = {};
#pragma unroll
    for (int w = 0; w < NWAVE; ++w) {
      G += *(const v8f*)&red[(w * 4 + ag) * 256 + lane * 8];
      U += *(const v8f*)&red[(w * 4 + au) * 256 + lane * 8];
    }
    // SwiGLU. C layout: VGPR r, lanes 0-15 -> M=r, lanes 16-31 -> M=r+8.
    const int mbase = wv * 16 + ((lane < 16) ? 0 : 8);
#pragma unroll
    for (int r = 0; r < 8; ++r) {
      float g = G[r], u = U[r];
      float a = (g / (1.0f + __expf(-g))) * u;
      act[(mbase + r) * KI_ + nt * 16 + ncol] = (_Float16)a;
    }
  }
}

// Kernel 2: out = act @ dequant(W_down) -> f16 [32, KH_].
// grid = KH_/16 = 256 -> 1024 concurrent waves with the 4-way K-split.
__global__ __launch_bounds__(32 * NWAVE)
void fp4_down_kernel(const _Float16* __restrict__ act,
                     const int* __restrict__ wpk,
                     const _Float16* __restrict__ wsc,
                     _Float16* __restrict__ out) {
  __shared__ float red[NWAVE * 2 * 256];  // 8 KB

  const int tid  = threadIdx.x;
  const int wv   = tid >> 5;
  const int lane = tid & 31;
  const int nt   = blockIdx.x;
  const int nb   = nt * 16;

  const int ncol = lane & 15;
  const int m    = lane & 15;
  const int koff = (lane < 16) ? 0 : 8;
  const int kwb  = (lane < 16) ? 0 : 1;

  v8f acc0 = {}, acc1 = {};

  const int tsteps = KI_ / 32 / NWAVE;  // 16 K-steps per wave
  const int t0 = wv * tsteps, t1 = t0 + tsteps;
  for (int t = t0; t < t1; ++t) {
    const int kb   = t * 32;
    const int rowp = t * 4 + kwb;

    unsigned int w0 = (unsigned int)__builtin_nontemporal_load(
        &wpk[(rowp    ) * KH_ + nb + ncol]);
    unsigned int w1 = (unsigned int)__builtin_nontemporal_load(
        &wpk[(rowp + 2) * KH_ + nb + ncol]);
    const _Float16 s = (_Float16)((float)wsc[(t >> 2) * KH_ + nb + ncol] * 16384.0f);
    if (t + 1 < t1) {
      __builtin_prefetch(&wpk[(rowp + 4) * KH_ + nb + ncol], 0, 1);
    }

    const _Float16* ar0 = act + m * KI_ + kb + koff;
    const _Float16* ar1 = act + (m + 16) * KI_ + kb + koff;
    const v16h A0 = cat16(*(const v8h*)ar0, *(const v8h*)(ar0 + 16));
    const v16h A1 = cat16(*(const v8h*)ar1, *(const v8h*)(ar1 + 16));

    const v16h Bm = cat16(dequant8(w0, s), dequant8(w1, s));

    acc0 = __builtin_amdgcn_wmma_f32_16x16x32_f16(false, A0, false, Bm,
                                                  (short)0, acc0, false, false);
    acc1 = __builtin_amdgcn_wmma_f32_16x16x32_f16(false, A1, false, Bm,
                                                  (short)0, acc1, false, false);
  }

  *(v8f*)&red[(wv * 2 + 0) * 256 + lane * 8] = acc0;
  *(v8f*)&red[(wv * 2 + 1) * 256 + lane * 8] = acc1;
  __syncthreads();

  if (wv < 2) {
    v8f A = {};
#pragma unroll
    for (int w = 0; w < NWAVE; ++w) {
      A += *(const v8f*)&red[(w * 2 + wv) * 256 + lane * 8];
    }
    const int mbase = wv * 16 + ((lane < 16) ? 0 : 8);
#pragma unroll
    for (int r = 0; r < 8; ++r) {
      out[(mbase + r) * KH_ + nb + ncol] = (_Float16)A[r];
    }
  }
}

extern "C" void kernel_launch(void* const* d_in, const int* in_sizes, int n_in,
                              void* d_out, int out_size, void* d_ws, size_t ws_size,
                              hipStream_t stream) {
  (void)in_sizes; (void)n_in; (void)out_size; (void)ws_size;
  const _Float16* x    = (const _Float16*)d_in[0];  // [32, 4096] f16
  const int*      gup  = (const int*)d_in[1];       // [512, 4096] packed FP4
  const _Float16* gus  = (const _Float16*)d_in[2];  // [32, 4096] f16 scales
  const int*      dwp  = (const int*)d_in[3];       // [256, 4096] packed FP4
  const _Float16* dws  = (const _Float16*)d_in[4];  // [16, 4096] f16 scales
  // d_in[5] = group_size scalar (=128, baked into the tiling)

  _Float16* outp  = (_Float16*)d_out;  // [32, 4096] f16
  _Float16* actws = (_Float16*)d_ws;   // [32, 2048] f16 intermediate (128 KB)

  fp4_gateup_swiglu_kernel<<<KI_ / 16, 32 * NWAVE, 0, stream>>>(x, gup, gus, actws);
  fp4_down_kernel<<<KH_ / 16, 32 * NWAVE, 0, stream>>>(actws, dwp, dws, outp);
}
